// BipolarDenseLayer_63900523430275
// MI455X (gfx1250) — matrix operand order repaired and checked by reading, not compiled, not run on hardware
//
#include <hip/hip_runtime.h>
#include <hip/hip_bf16.h>
#include <stdint.h>
#include <stddef.h>

// Problem dims (match reference)
#define MDIM 8192
#define KDIM 4096
#define NDIM 4096

// GEMM tiling
#define BT_M 128
#define BT_N 128
#define BT_K 32
#define LDSROW 40              // padded row pitch (bf16): 80B = 20-bank stride, conflict-free b128
#define NKSTEPS (KDIM / BT_K)  // 128

typedef __bf16 bf16;
typedef __attribute__((ext_vector_type(16))) __bf16 v16bf;
typedef __attribute__((ext_vector_type(8)))  __bf16 v8bf;
typedef __attribute__((ext_vector_type(4)))  __bf16 v4bf;
typedef __attribute__((ext_vector_type(8)))  float  v8f;
typedef __attribute__((ext_vector_type(4)))  float  v4f;
typedef __attribute__((ext_vector_type(4)))  uint32_t su32x4;
typedef __attribute__((ext_vector_type(8)))  uint32_t su32x8;

__device__ __forceinline__ v16bf combine8(v8bf c0, v8bf c1) {
    v16bf r;
#pragma unroll
    for (int i = 0; i < 8; ++i) { r[i] = c0[i]; r[8 + i] = c1[i]; }
    return r;
}

// Generic pointer -> 32-bit LDS byte address (AS3 pointers are 32-bit offsets)
__device__ __forceinline__ uint32_t lds_byte_addr(const void* p) {
    return (uint32_t)(uintptr_t)(const __attribute__((address_space(3))) void*)p;
}

// ---------------------------------------------------------------------------
// TDM: load one 128(rows) x 32(K, contiguous) bf16 tile into LDS with the
// hardware padding 64B data + 16B pad per row => 80B LDS row pitch (LDSROW).
// 2D tensor => 2 SGPR descriptor groups (VADDR2/3 = NULL form).
// ---------------------------------------------------------------------------
__device__ __forceinline__ void tdm_load_tile(const bf16* gsrc, uint32_t lds_addr) {
    const uint64_t ga = (uint64_t)(uintptr_t)gsrc;
    su32x4 g0;
    g0[0] = 1u;                                     // count = 1 (valid user D#)
    g0[1] = lds_addr;                               // LDS dest byte address
    g0[2] = (uint32_t)ga;                           // global_addr[31:0]
    g0[3] = ((uint32_t)(ga >> 32) & 0x01FFFFFFu)    // global_addr[56:32]
          | (2u << 30);                             // type = 2 ("image")
    su32x8 g1;
    g1[0] = (1u << 16)                              // data_size = 1 -> 2 bytes
          | (1u << 20)                              // pad_enable
          | (3u << 22)                              // pad_interval: 16 DWORDs (64B row)
          | (3u << 25);                             // pad_amount: 4 DWORDs (16B pad)
    g1[1] = ((uint32_t)KDIM & 0xFFFFu) << 16;       // tensor_dim0 lo16 (=4096)
    g1[2] = ((uint32_t)KDIM >> 16)                  // tensor_dim0 hi16
          | ((uint32_t)BT_N << 16);                 // tensor_dim1 lo16 (=128 rows)
    g1[3] = ((uint32_t)BT_K << 16);                 // tensor_dim1 hi16 | tile_dim0 = 32
    g1[4] = (uint32_t)BT_N;                         // tile_dim1 = 128, tile_dim2 = 0
    g1[5] = (uint32_t)KDIM;                         // tensor_dim0_stride lo32 (=4096)
    g1[6] = 0u;                                     // stride0 hi16 | stride1 lo16
    g1[7] = 0u;                                     // stride1 hi32
    asm volatile("tensor_load_to_lds %0, %1" :: "s"(g0), "s"(g1) : "memory");
}

// ---------------------------------------------------------------------------
// Kernel 1: swT[n][k] = bf16(sign(w[k][n]))  (tiled transpose, coalesced R/W)
// ---------------------------------------------------------------------------
__global__ __launch_bounds__(256) void pack_signw(const float* __restrict__ w,
                                                  bf16* __restrict__ swT) {
    __shared__ float tile[32][33];
    const int tx = threadIdx.x;  // 0..31
    const int ty = threadIdx.y;  // 0..7
    const int n0 = blockIdx.x * 32;
    const int k0 = blockIdx.y * 32;
#pragma unroll
    for (int i = 0; i < 4; ++i) {
        const int k = ty + i * 8;
        tile[k][tx] = w[(size_t)(k0 + k) * NDIM + (n0 + tx)];
    }
    __syncthreads();
#pragma unroll
    for (int i = 0; i < 4; ++i) {
        const int n = ty + i * 8;
        const float v = tile[tx][n];
        const float s = (v > 0.0f) ? 1.0f : ((v < 0.0f) ? -1.0f : 0.0f);
        swT[(size_t)(n0 + n) * KDIM + (k0 + tx)] = (bf16)s;  // exact in bf16
    }
}

// ---------------------------------------------------------------------------
// Kernel 2: split x (fp32) into bf16 hi + residual lo once (near-fp32 GEMM)
// ---------------------------------------------------------------------------
__global__ __launch_bounds__(256) void split_x(const float* __restrict__ x,
                                               bf16* __restrict__ xhi,
                                               bf16* __restrict__ xlo) {
    const size_t i4 = ((size_t)blockIdx.x * 256 + threadIdx.x) * 4;
    const v4f v = *(const v4f*)(x + i4);
    v4bf h, l;
#pragma unroll
    for (int j = 0; j < 4; ++j) {
        const float f  = v[j];
        const bf16  hh = (bf16)f;          // RNE
        h[j] = hh;
        l[j] = (bf16)(f - (float)hh);      // residual
    }
    *(v4bf*)(xhi + i4) = h;
    *(v4bf*)(xlo + i4) = l;
}

// ---------------------------------------------------------------------------
// Kernel 3: out = relu((xhi+xlo) @ signW + b); all tiles streamed by TDM,
// double-buffered LDS, 8 wave32 per block, 2x4 16x16 WMMA tiles per wave.
// ---------------------------------------------------------------------------
__global__ __launch_bounds__(256) void bipolar_gemm(const bf16* __restrict__ xhi,
                                                    const bf16* __restrict__ xlo,
                                                    const bf16* __restrict__ swT,
                                                    const float* __restrict__ bias,
                                                    float* __restrict__ out) {
    __shared__ __align__(16) bf16 sAhi[2][BT_M * LDSROW];
    __shared__ __align__(16) bf16 sAlo[2][BT_M * LDSROW];
    __shared__ __align__(16) bf16 sB  [2][BT_N * LDSROW];

    const int tid   = threadIdx.x;
    const int lane  = tid & 31;
    const int wave  = tid >> 5;
    const int waveM = wave >> 1;     // 0..3 -> 32-row strip
    const int waveN = wave & 1;      // 0..1 -> 64-col strip
    const int ml    = lane & 15;
    const bool loh  = (lane < 16);
    const int kofsA = loh ? 0 : 8;   // 16-bit A frag: lanes 0-15 K{0-7,16-23}, 16-31 K{8-15,24-31}
    const int kofsB = loh ? 0 : 8;   // 16-bit B frag: lanes 0-15 K{0-7,16-23}, 16-31 K{8-15,24-31}

    const int m0 = blockIdx.y * BT_M;
    const int n0 = blockIdx.x * BT_N;

    // Scalar leader condition: whole-wave skip for waves 1..7 via s_cbranch
    const bool leader = (__builtin_amdgcn_readfirstlane(tid) < 32);

    v8f acc[2][4] = {};

    auto tdm_stage = [&](int kt, int st) {  // leader wave only
        const size_t koff = (size_t)kt * BT_K;
        tdm_load_tile(xhi + (size_t)m0 * KDIM + koff, lds_byte_addr(&sAhi[st][0]));
        tdm_load_tile(xlo + (size_t)m0 * KDIM + koff, lds_byte_addr(&sAlo[st][0]));
        tdm_load_tile(swT + (size_t)n0 * KDIM + koff, lds_byte_addr(&sB[st][0]));
    };

    auto compute = [&](int st) {
        v16bf ahi[2], alo[2];
#pragma unroll
        for (int sm = 0; sm < 2; ++sm) {
            const bf16* ap = &sAhi[st][(waveM * 32 + sm * 16 + ml) * LDSROW + kofsA];
            ahi[sm] = combine8(*(const v8bf*)ap, *(const v8bf*)(ap + 16));
            const bf16* lp = &sAlo[st][(waveM * 32 + sm * 16 + ml) * LDSROW + kofsA];
            alo[sm] = combine8(*(const v8bf*)lp, *(const v8bf*)(lp + 16));
        }
#pragma unroll
        for (int sn = 0; sn < 4; ++sn) {
            const bf16* bp = &sB[st][(waveN * 64 + sn * 16 + ml) * LDSROW + kofsB];
            const v16bf bfrag = combine8(*(const v8bf*)bp, *(const v8bf*)(bp + 16));
#pragma unroll
            for (int sm = 0; sm < 2; ++sm) {
                acc[sm][sn] = __builtin_amdgcn_wmma_f32_16x16x32_bf16(
                    false, ahi[sm], false, bfrag, (short)0, acc[sm][sn], false, false);
                acc[sm][sn] = __builtin_amdgcn_wmma_f32_16x16x32_bf16(
                    false, alo[sm], false, bfrag, (short)0, acc[sm][sn], false, false);
            }
        }
    };

    // Prologue: fill stage 0 via TDM
    if (leader) {
        tdm_stage(0, 0);
        __builtin_amdgcn_s_wait_tensorcnt(0);
    }
    __syncthreads();

    for (int kt = 0; kt < NKSTEPS; ++kt) {
        const int st = kt & 1;
        if (leader && (kt + 1 < NKSTEPS)) tdm_stage(kt + 1, st ^ 1);  // overlap with compute
        compute(st);
        if (leader) __builtin_amdgcn_s_wait_tensorcnt(0);             // next stage landed
        __syncthreads();
    }

    // Epilogue: C layout -> VGPR r holds row r (lanes 0-15) / row r+8 (lanes 16-31)
#pragma unroll
    for (int sm = 0; sm < 2; ++sm) {
#pragma unroll
        for (int sn = 0; sn < 4; ++sn) {
            const int row0 = m0 + waveM * 32 + sm * 16 + (loh ? 0 : 8);
            const int col  = n0 + waveN * 64 + sn * 16 + ml;
            const float bv = bias[col];
#pragma unroll
            for (int r = 0; r < 8; ++r) {
                const float v = acc[sm][sn][r] + bv;
                out[(size_t)(row0 + r) * NDIM + col] = (v > 0.0f) ? v : 0.0f;
            }
        }
    }
}

// ---------------------------------------------------------------------------
extern "C" void kernel_launch(void* const* d_in, const int* in_sizes, int n_in,
                              void* d_out, int out_size, void* d_ws, size_t ws_size,
                              hipStream_t stream) {
    const float* x  = (const float*)d_in[0];
    const float* w  = (const float*)d_in[1];
    const float* bb = (const float*)d_in[2];
    float* out = (float*)d_out;

    // Workspace layout: swT (32 MiB) | xhi (64 MiB) | xlo (64 MiB)
    const size_t swSZ = (size_t)KDIM * NDIM * sizeof(bf16);
    const size_t xSZ  = (size_t)MDIM * KDIM * sizeof(bf16);
    bf16* swT = (bf16*)d_ws;
    bf16* xhi = (bf16*)((char*)d_ws + swSZ);
    bf16* xlo = (bf16*)((char*)d_ws + swSZ + xSZ);

    {
        dim3 grid(NDIM / 32, KDIM / 32);
        dim3 block(32, 8);
        pack_signw<<<grid, block, 0, stream>>>(w, swT);
    }
    {
        const int blocks = (int)((size_t)MDIM * KDIM / 4 / 256);  // 32768
        split_x<<<blocks, 256, 0, stream>>>(x, xhi, xlo);
    }
    {
        dim3 grid(NDIM / BT_N, MDIM / BT_M);  // (32, 64)
        dim3 block(256);
        bipolar_gemm<<<grid, block, 0, stream>>>(xhi, xlo, swT, bb, out);
    }
}
